// GatedDeltaNet_30674656428351
// MI455X (gfx1250) — matrix operand order repaired and checked by reading, not compiled, not run on hardware
//
#include <hip/hip_runtime.h>

// ---------------------------------------------------------------------------
// GatedDeltaNet forward for MI455X (gfx1250, wave32, WMMA + async-to-LDS).
//
// Math note: a*(S + b*v k^T) + (1-a)*S == S + (a*b)*v k^T, so the scan is a
// cumulative sum of rank-1 updates -> chunked linear attention:
//   o_t = Q_t @ S_prev(chunk)^T + tril(Q K^T) diag(ab) V   (per 64-token chunk)
// All big GEMMs (x@Wq/Wk/Wv, o@Wo, proj@Wg: ~120 GFLOP) run through
// v_wmma_f32_16x16x32_bf16; tiles are staged global->LDS with
// GLOBAL_LOAD_ASYNC_TO_LDS_B128 (ASYNCcnt) and double-buffered.
// Workspace requirement: ~287 MB (buffers overlaid after their last use).
// ---------------------------------------------------------------------------

typedef unsigned short u16;
typedef __attribute__((ext_vector_type(8)))  u16    v8u;
typedef __attribute__((ext_vector_type(16))) u16    v16u;
typedef __attribute__((ext_vector_type(16))) __bf16 v16bf;
typedef __attribute__((ext_vector_type(8)))  float  v8f;
typedef int v4i __attribute__((vector_size(16)));   // matches builtin's V4i

constexpr int Bc   = 4;
constexpr int Tc   = 2048;
constexpr int DIMc = 1024;
constexpr int Hc   = 16;
constexpr int HDc  = 64;
constexpr int KC   = 4;                 // conv taps
constexpr int QKc  = Hc * HDc;          // 1024
constexpr int VDc  = 2 * Hc * HDc;      // 2048
constexpr int Mc   = Bc * Tc;           // 8192
constexpr int CHUNK = 64;
constexpr int NCH   = Tc / CHUNK;       // 32
constexpr int BH    = Bc * Hc;          // 64
constexpr int SSZ   = 2 * HDc * HDc;    // 8192 state elems per (b,h)
constexpr float EPSc = 1e-5f;

__device__ __forceinline__ u16 f2bf(float f) {
  union { float f; unsigned u; } x; x.f = f;
  unsigned r = x.u + 0x7FFFu + ((x.u >> 16) & 1u);   // round-to-nearest-even
  return (u16)(r >> 16);
}
__device__ __forceinline__ float sigmoidf(float v) { return 1.0f / (1.0f + __expf(-v)); }

// ---------------------------------------------------------------------------
// CDNA5 async global->LDS copy (tracked by ASYNCcnt), guarded so an older
// toolchain falls back to the synchronous VGPR round-trip path.
// AS3 pointer = low 32 bits of the generic pointer (addrspacecast generic->
// local on AMDGPU is a 32-bit truncation); AS1 pointer = same 64-bit VA.
// Builtin signature (from hipcc diagnostic): (int4 AS1*, int4 AS3*, Ii, Ii).
// ---------------------------------------------------------------------------
#if __has_builtin(__builtin_amdgcn_global_load_async_to_lds_b128) && \
    __has_builtin(__builtin_amdgcn_s_wait_asynccnt)
#define USE_ASYNC_LDS 1
#else
#define USE_ASYNC_LDS 0
#endif

#define AS_GLOBAL __attribute__((address_space(1)))
#define AS_LOCAL  __attribute__((address_space(3)))

__device__ __forceinline__ void async_b128(const void* g, void* l) {
#if USE_ASYNC_LDS
  __builtin_amdgcn_global_load_async_to_lds_b128(
      (AS_GLOBAL v4i*)(unsigned long long)(uintptr_t)g,
      (AS_LOCAL  v4i*)(unsigned int)(uintptr_t)l,
      0, 0);
#else
  *(v8u*)l = *(const v8u*)g;   // sync fallback: global_load + ds_store
#endif
}
__device__ __forceinline__ void wait_async0() {
#if USE_ASYNC_LDS
  __builtin_amdgcn_s_wait_asynccnt(0);
#endif
}

// ---------------------------------------------------------------------------
// Elementwise casts
// ---------------------------------------------------------------------------
__global__ void cast_bf16_k(const float* __restrict__ in, u16* __restrict__ out, size_t n) {
  size_t stride = (size_t)gridDim.x * blockDim.x;
  for (size_t i = (size_t)blockIdx.x * blockDim.x + threadIdx.x; i < n; i += stride)
    out[i] = f2bf(in[i]);
}

// in: [rows, cols] f32  ->  out: [cols, rows] bf16   (weights -> [N,K] layout)
__global__ void transpose_cast_k(const float* __restrict__ in, u16* __restrict__ out,
                                 int rows, int cols) {
  size_t n = (size_t)rows * cols;
  size_t stride = (size_t)gridDim.x * blockDim.x;
  for (size_t i = (size_t)blockIdx.x * blockDim.x + threadIdx.x; i < n; i += stride) {
    size_t r = i / cols, c = i % cols;
    out[c * (size_t)rows + r] = f2bf(in[i]);
  }
}

// ---------------------------------------------------------------------------
// WMMA bf16 GEMM: C[M,N] f32 = A[M,K] bf16(row-major) * Bt[N,K] bf16
// block = 256 threads (8 waves), tile 128(M) x 256(N), K-step 32.
// Wave grid 2x4; each wave owns 64x64 -> 16 WMMAs per K-step per wave.
// Double-buffered LDS, async global->LDS staging overlapped with compute.
// ---------------------------------------------------------------------------
constexpr int PITCH = 40;   // padded LDS row pitch in halves (32 used + 8)

__global__ __launch_bounds__(256)
void wmma_gemm_bf16(const u16* __restrict__ A, const u16* __restrict__ Bt,
                    float* __restrict__ C, int Mdim, int Ndim, int Kdim) {
  __shared__ u16 As[2][128 * PITCH];   // 2 x 10 KB
  __shared__ u16 Bs[2][256 * PITCH];   // 2 x 20 KB (B transposed: [n][k])

  const int mBase = blockIdx.y * 128;
  const int nBase = blockIdx.x * 256;
  const int tid  = threadIdx.x;
  const int wave = tid >> 5;
  const int lane = tid & 31;
  const int wm   = (wave >> 2) * 64;   // 0 / 64
  const int wn   = (wave & 3) * 64;    // 0,64,128,192
  const int lrow = lane & 15;
  const int half = lane >> 4;          // 0: lanes 0-15, 1: lanes 16-31

  v8f acc[4][4];
#pragma unroll
  for (int i = 0; i < 4; ++i)
#pragma unroll
    for (int j = 0; j < 4; ++j) acc[i][j] = (v8f)0.0f;

  // per-thread staging slices
  const int rA = tid >> 1;            // 0..127 : A tile row
  const int cA = (tid & 1) * 16;      // 0 / 16 : half-row of 32 halves
  const u16* aSrc = A  + (size_t)(mBase + rA) * Kdim + cA;
  const u16* bSrc = Bt + (size_t)(nBase + tid) * Kdim;   // one B row per thread

  auto issue_tile = [&](int kt, int buf) {
    const u16* as = aSrc + kt;
    u16* ad = &As[buf][rA * PITCH + cA];
    async_b128(as,     ad);
    async_b128(as + 8, ad + 8);
    const u16* bs = bSrc + kt;
    u16* bd = &Bs[buf][tid * PITCH];
    async_b128(bs,      bd);
    async_b128(bs + 8,  bd + 8);
    async_b128(bs + 16, bd + 16);
    async_b128(bs + 24, bd + 24);
  };

  issue_tile(0, 0);
  const int nk = Kdim >> 5;
  for (int it = 0; it < nk; ++it) {
    const int cur = it & 1;
    wait_async0();        // my tile-it loads done
    __syncthreads();      // everyone's loads visible; prev reads of buf^1 done
    if (it + 1 < nk) issue_tile((it + 1) << 5, cur ^ 1);

    // fragments per the ISA 16-bit A(16x32)/B(32x16) lane maps
    v16bf afr[4], bfr[4];
#pragma unroll
    for (int i = 0; i < 4; ++i) {
      // lane half 0: K 0..7 & 16..23 ; half 1: K 8..15 & 24..31 (row = M)
      const u16* p = &As[cur][(wm + i * 16 + lrow) * PITCH + half * 8];
      v8u lo = *(const v8u*)p;
      v8u hi = *(const v8u*)(p + 16);
      v16u w = __builtin_shufflevector(lo, hi, 0,1,2,3,4,5,6,7,8,9,10,11,12,13,14,15);
      afr[i] = __builtin_bit_cast(v16bf, w);
    }
#pragma unroll
    for (int j = 0; j < 4; ++j) {
      // lane half 0: K 0..7 & 16..23 ; half 1: K 8..15 & 24..31 (col = N)
      const u16* p = &Bs[cur][(wn + j * 16 + lrow) * PITCH + half * 8];
      v8u lo = *(const v8u*)p;
      v8u hi = *(const v8u*)(p + 16);
      v16u w = __builtin_shufflevector(lo, hi, 0,1,2,3,4,5,6,7,8,9,10,11,12,13,14,15);
      bfr[j] = __builtin_bit_cast(v16bf, w);
    }

#pragma unroll
    for (int i = 0; i < 4; ++i)
#pragma unroll
      for (int j = 0; j < 4; ++j)
        acc[i][j] = __builtin_amdgcn_wmma_f32_16x16x32_bf16(
            false, afr[i], false, bfr[j], (short)0, acc[i][j], false, false);
    __syncthreads();
  }

  // store: C/D layout: VGPR rr -> M = rr (+8 for lanes 16-31), N = lane&15
#pragma unroll
  for (int i = 0; i < 4; ++i) {
#pragma unroll
    for (int j = 0; j < 4; ++j) {
      int n = nBase + wn + j * 16 + lrow;
#pragma unroll
      for (int rr = 0; rr < 8; ++rr) {
        int m = mBase + wm + i * 16 + half * 8 + rr;
        C[(size_t)m * Ndim + n] = acc[i][j][rr];
      }
    }
  }
}

// ---------------------------------------------------------------------------
// Zero-centered RMS norm, in place, one block per token row.
// ---------------------------------------------------------------------------
__global__ __launch_bounds__(256)
void zrms_kernel(float* __restrict__ y, const float* __restrict__ g, int C) {
  __shared__ float rs[256], rss[256];
  float* p = y + (size_t)blockIdx.x * C;
  int tid = threadIdx.x;
  float s = 0.f, ss = 0.f;
  for (int c = tid; c < C; c += 256) { float v = p[c]; s += v; ss += v * v; }
  rs[tid] = s; rss[tid] = ss;
  __syncthreads();
  for (int st = 128; st > 0; st >>= 1) {
    if (tid < st) { rs[tid] += rs[tid + st]; rss[tid] += rss[tid + st]; }
    __syncthreads();
  }
  float mean  = rs[0] / C;
  float scale = rsqrtf(rss[0] / C - mean * mean + EPSc);
  for (int c = tid; c < C; c += 256) p[c] = (p[c] - mean) * scale * g[c];
}

// ---------------------------------------------------------------------------
// Causal depthwise conv (K=4) + SiLU, elementwise grid-stride.
// ---------------------------------------------------------------------------
__global__ void dwconv_silu(const float* __restrict__ in, const float* __restrict__ w,
                            float* __restrict__ out, int C, size_t total) {
  size_t stride = (size_t)gridDim.x * blockDim.x;
  for (size_t i = (size_t)blockIdx.x * blockDim.x + threadIdx.x; i < total; i += stride) {
    int c = (int)(i % C);
    int t = (int)((i / C) % Tc);
    float acc = 0.f;
#pragma unroll
    for (int j = 0; j < KC; ++j) {
      int ts = t - (KC - 1) + j;
      if (ts >= 0)
        acc += w[j * C + c] * in[(long long)i + (long long)(ts - t) * C];
    }
    out[i] = acc * sigmoidf(acc);
  }
}

// ---------------------------------------------------------------------------
// c[b,t,h] = sigmoid(x@Wa) * sigmoid(x@Wb)   (the collapsed gate a*b)
// ---------------------------------------------------------------------------
__global__ __launch_bounds__(256)
void gate_ab(const float* __restrict__ x, const float* __restrict__ Wa,
             const float* __restrict__ Wb, float* __restrict__ cbuf) {
  __shared__ float xs[DIMc];
  __shared__ float part[256];
  __shared__ float outv[32];
  size_t row = blockIdx.x;
  int tid = threadIdx.x;
  for (int c = tid; c < DIMc; c += 256) xs[c] = x[row * DIMc + c];
  __syncthreads();
  int o   = tid & 31;           // 0..15 -> Wa, 16..31 -> Wb
  int seg = tid >> 5;           // 8 K-segments of 128
  const float* W = (o < 16) ? Wa : Wb;
  int hcol = o & 15;
  float p = 0.f;
  for (int kk = seg * 128; kk < seg * 128 + 128; ++kk) p += xs[kk] * W[kk * Hc + hcol];
  part[tid] = p;
  __syncthreads();
  if (tid < 32) {
    float s = 0.f;
    for (int rseg = 0; rseg < 8; ++rseg) s += part[rseg * 32 + tid];
    outv[tid] = s;
  }
  __syncthreads();
  if (tid < 16)
    cbuf[row * Hc + tid] = sigmoidf(outv[tid]) * sigmoidf(outv[16 + tid]);
}

// ---------------------------------------------------------------------------
// Per-chunk rank-1 state deltas: D[bh,chunk] = sum_{s in chunk} c_s v_s k_s^T
// ---------------------------------------------------------------------------
__global__ __launch_bounds__(256)
void chunk_delta(const float* __restrict__ kbuf, const float* __restrict__ vbuf,
                 const float* __restrict__ cbuf, float* __restrict__ D) {
  __shared__ float ks[HDc];
  __shared__ float vs[2 * HDc];
  __shared__ float cs;
  int chunk = blockIdx.x, bh = blockIdx.y;
  int b = bh >> 4, h = bh & 15;
  int t0 = chunk * CHUNK;
  int tid = threadIdx.x;
  int v  = tid & 127;
  int kh = tid >> 7;            // 0/1 -> K half
  float acc[32];
#pragma unroll
  for (int i = 0; i < 32; ++i) acc[i] = 0.f;

  for (int s = 0; s < CHUNK; ++s) {
    size_t t = (size_t)b * Tc + t0 + s;
    __syncthreads();
    if (tid < 64)        ks[tid]       = kbuf[t * QKc + h * HDc + tid];
    else if (tid < 192)  vs[tid - 64]  = vbuf[t * VDc + h * 2 * HDc + (tid - 64)];
    else if (tid == 192) cs            = cbuf[t * Hc + h];
    __syncthreads();
    float cv = cs * vs[v];
#pragma unroll
    for (int kk = 0; kk < 32; ++kk) acc[kk] += cv * ks[kh * 32 + kk];
  }
  float* dst = D + ((size_t)bh * NCH + chunk) * SSZ + (size_t)v * HDc + kh * 32;
#pragma unroll
  for (int kk = 0; kk < 32; ++kk) dst[kk] = acc[kk];
}

// ---------------------------------------------------------------------------
// In-place exclusive prefix over the 32 chunk states per (b,h).
// ---------------------------------------------------------------------------
__global__ __launch_bounds__(256)
void state_scan(float* __restrict__ D) {
  size_t base = (size_t)blockIdx.x * NCH * SSZ + (size_t)threadIdx.x * 32;
  float run[32];
#pragma unroll
  for (int i = 0; i < 32; ++i) run[i] = 0.f;
  for (int c = 0; c < NCH; ++c) {
    float* p = D + base + (size_t)c * SSZ;
#pragma unroll
    for (int i = 0; i < 32; ++i) { float t = p[i]; p[i] = run[i]; run[i] += t; }
  }
}

// ---------------------------------------------------------------------------
// Chunk output: O = Q @ Sprev^T + tril(Q K^T) diag(c) V      (per (bh,chunk))
// LDS aliased: Xs holds K during attention, then Sprev. ~98 KB total.
// ---------------------------------------------------------------------------
__global__ __launch_bounds__(256)
void chunk_output(const float* __restrict__ qbuf, const float* __restrict__ kbuf,
                  const float* __restrict__ vbuf, const float* __restrict__ cbuf,
                  const float* __restrict__ Sprev, float* __restrict__ obuf) {
  __shared__ float smem[4160 + 8256 + 4160 + 64 + 8320];
  float* Qs  = smem;            // [64][65]
  float* Vs  = Qs  + 4160;      // [64][129]
  float* Am  = Vs  + 8256;      // [64][65]  attention (masked, *c)
  float* Cs  = Am  + 4160;      // [64]
  float* Xs  = Cs  + 64;        // phase1: K [64][65]; phase2: Sprev [128][65]

  int chunk = blockIdx.x, bh = blockIdx.y;
  int b = bh >> 4, h = bh & 15;
  int t0 = chunk * CHUNK;
  int tid = threadIdx.x;

  for (int e = tid; e < 64 * 64; e += 256) {
    int tt = e >> 6, k = e & 63;
    size_t gt = (size_t)b * Tc + t0 + tt;
    Qs[tt * 65 + k] = qbuf[gt * QKc + h * HDc + k];
    Xs[tt * 65 + k] = kbuf[gt * QKc + h * HDc + k];
  }
  for (int e = tid; e < 64 * 128; e += 256) {
    int tt = e >> 7, v = e & 127;
    Vs[tt * 129 + v] = vbuf[((size_t)b * Tc + t0 + tt) * VDc + h * 128 + v];
  }
  if (tid < 64) Cs[tid] = cbuf[((size_t)b * Tc + t0 + tid) * Hc + h];
  __syncthreads();

  // attention scores (causal, scaled by c_s)
  for (int e = tid; e < 64 * 64; e += 256) {
    int tt = e >> 6, s = e & 63;
    float a = 0.f;
    if (s <= tt) {
#pragma unroll 8
      for (int k = 0; k < 64; ++k) a += Qs[tt * 65 + k] * Xs[s * 65 + k];
      a *= Cs[s];
    }
    Am[tt * 65 + s] = a;
  }
  __syncthreads();

  // swap K -> Sprev in the aliased region
  {
    const float* sp = Sprev + ((size_t)bh * NCH + chunk) * SSZ;
    for (int e = tid; e < SSZ; e += 256) Xs[(e >> 6) * 65 + (e & 63)] = sp[e];
  }
  __syncthreads();

  int tt = tid >> 2;
  int v0 = (tid & 3) * 32;
  float res[32];
#pragma unroll
  for (int i = 0; i < 32; ++i) res[i] = 0.f;
  for (int s = 0; s < 64; ++s) {
    float a = Am[tt * 65 + s];
#pragma unroll
    for (int i = 0; i < 32; ++i) res[i] += a * Vs[s * 129 + v0 + i];
  }
  for (int k = 0; k < 64; ++k) {
    float qv = Qs[tt * 65 + k];
#pragma unroll
    for (int i = 0; i < 32; ++i) res[i] += qv * Xs[(v0 + i) * 65 + k];
  }
  float* op = obuf + ((size_t)b * Tc + t0 + tt) * VDc + h * 128 + v0;
#pragma unroll
  for (int i = 0; i < 32; ++i) op[i] = res[i];
}

// ---------------------------------------------------------------------------
// out = sigmoid(g) * proj
// ---------------------------------------------------------------------------
__global__ void gate_out_k(const float* __restrict__ proj, const float* __restrict__ g,
                           float* __restrict__ out, size_t n) {
  size_t stride = (size_t)gridDim.x * blockDim.x;
  for (size_t i = (size_t)blockIdx.x * blockDim.x + threadIdx.x; i < n; i += stride)
    out[i] = sigmoidf(g[i]) * proj[i];
}

// ---------------------------------------------------------------------------
extern "C" void kernel_launch(void* const* d_in, const int* in_sizes, int n_in,
                              void* d_out, int out_size, void* d_ws, size_t ws_size,
                              hipStream_t stream) {
  (void)in_sizes; (void)n_in; (void)out_size; (void)ws_size;
  const float* x  = (const float*)d_in[0];
  const float* Wq = (const float*)d_in[1];
  const float* Wk = (const float*)d_in[2];
  const float* Wv = (const float*)d_in[3];
  const float* gq = (const float*)d_in[4];
  const float* gk = (const float*)d_in[5];
  const float* gv = (const float*)d_in[6];
  const float* cq = (const float*)d_in[7];
  const float* ck = (const float*)d_in[8];
  const float* cv = (const float*)d_in[9];
  const float* Wa = (const float*)d_in[10];
  const float* Wb = (const float*)d_in[11];
  const float* Wo = (const float*)d_in[12];
  const float* Wg = (const float*)d_in[13];
  float* out = (float*)d_out;
  char*  ws  = (char*)d_ws;

  constexpr size_t MB = 1ull << 20;
  // ---- workspace layout (peak 287 MB, with overlays after last use) ----
  u16*   xb   = (u16*)  (ws + 0);          // 16 MB  x bf16
  u16*   wqT  = (u16*)  (ws + 16 * MB);    //  2 MB
  u16*   wkT  = (u16*)  (ws + 18 * MB);    //  2 MB
  u16*   wvT  = (u16*)  (ws + 20 * MB);    //  4 MB
  u16*   woT  = (u16*)  (ws + 24 * MB);    //  4 MB
  u16*   wgT  = (u16*)  (ws + 28 * MB);    //  2 MB
  float* cbuf = (float*)(ws + 30 * MB);    //  0.5 MB  c = sig(a)*sig(b)
  float* qp   = (float*)(ws + 31 * MB);    // 32 MB  q proj / normed
  float* kp   = (float*)(ws + 63 * MB);    // 32 MB
  float* vp   = (float*)(ws + 95 * MB);    // 64 MB
  float* qc   = (float*)(ws + 159 * MB);   // 32 MB  conv+silu outputs
  float* kc   = (float*)(ws + 191 * MB);   // 32 MB
  float* vc   = (float*)(ws + 223 * MB);   // 64 MB  (peak end: 287 MB)
  // overlays (inputs dead by the time these are written):
  float* D    = (float*)(ws + 31 * MB);    // 64 MB over qp+kp (chunk states)
  float* obuf = (float*)(ws + 95 * MB);    // 64 MB over vp    (attention out)
  u16*   obf  = (u16*)  (ws + 159 * MB);   // 32 MB over qc    (o bf16)
  float* proj = (float*)(ws + 191 * MB);   // 32 MB over kc
  u16*   pbf  = (u16*)  (ws + 31 * MB);    // 16 MB over D     (proj bf16)
  float* gbuf = (float*)(ws + 47 * MB);    // 32 MB over D     (proj@Wg)

  // 1) casts + weight transposes (one-time, tiny vs GEMMs)
  cast_bf16_k<<<2048, 256, 0, stream>>>(x, xb, (size_t)Mc * DIMc);
  transpose_cast_k<<<512, 256, 0, stream>>>(Wq, wqT, DIMc, QKc);
  transpose_cast_k<<<512, 256, 0, stream>>>(Wk, wkT, DIMc, QKc);
  transpose_cast_k<<<1024, 256, 0, stream>>>(Wv, wvT, DIMc, VDc);
  transpose_cast_k<<<1024, 256, 0, stream>>>(Wo, woT, VDc, DIMc);
  transpose_cast_k<<<512, 256, 0, stream>>>(Wg, wgT, DIMc, DIMc);

  // 2) projections (WMMA, N-tile 256)
  wmma_gemm_bf16<<<dim3(QKc / 256, Mc / 128), 256, 0, stream>>>(xb, wqT, qp, Mc, QKc, DIMc);
  wmma_gemm_bf16<<<dim3(QKc / 256, Mc / 128), 256, 0, stream>>>(xb, wkT, kp, Mc, QKc, DIMc);
  wmma_gemm_bf16<<<dim3(VDc / 256, Mc / 128), 256, 0, stream>>>(xb, wvT, vp, Mc, VDc, DIMc);

  // 3) gate c = sigmoid(x@Wa)*sigmoid(x@Wb)
  gate_ab<<<Mc, 256, 0, stream>>>(x, Wa, Wb, cbuf);

  // 4) zero-centered RMS norm (in place)
  zrms_kernel<<<Mc, 256, 0, stream>>>(qp, gq, QKc);
  zrms_kernel<<<Mc, 256, 0, stream>>>(kp, gk, QKc);
  zrms_kernel<<<Mc, 256, 0, stream>>>(vp, gv, VDc);

  // 5) causal depthwise conv + SiLU
  dwconv_silu<<<4096, 256, 0, stream>>>(qp, cq, qc, QKc, (size_t)Mc * QKc);
  dwconv_silu<<<4096, 256, 0, stream>>>(kp, ck, kc, QKc, (size_t)Mc * QKc);
  dwconv_silu<<<4096, 256, 0, stream>>>(vp, cv, vc, VDc, (size_t)Mc * VDc);

  // 6) chunked linear-attention recurrence (S_t = S_{t-1} + c_t v_t k_t^T)
  chunk_delta<<<dim3(NCH, BH), 256, 0, stream>>>(kc, vc, cbuf, D);
  state_scan<<<BH, 256, 0, stream>>>(D);
  chunk_output<<<dim3(NCH, BH), 256, 0, stream>>>(qc, kc, vc, cbuf, D, obuf);

  // 7) output projection + sigmoid gate (WMMA)
  cast_bf16_k<<<2048, 256, 0, stream>>>(obuf, obf, (size_t)Mc * VDc);
  wmma_gemm_bf16<<<dim3(DIMc / 256, Mc / 128), 256, 0, stream>>>(obf, woT, proj, Mc, DIMc, VDc);
  cast_bf16_k<<<2048, 256, 0, stream>>>(proj, pbf, (size_t)Mc * DIMc);
  wmma_gemm_bf16<<<dim3(DIMc / 256, Mc / 128), 256, 0, stream>>>(pbf, wgT, gbuf, Mc, DIMc, DIMc);
  gate_out_k<<<2048, 256, 0, stream>>>(proj, gbuf, out, (size_t)Mc * DIMc);
}
